// HeteroGAE_Encoder_18124761989368
// MI455X (gfx1250) — matrix operand order repaired and checked by reading, not compiled, not run on hardware
//
#include <hip/hip_runtime.h>

// ---------------------------------------------------------------------------
// HeteroGAE encoder for MI455X (gfx1250, wave32, WMMA).
// All dense projections run on V_WMMA_F32_16X16X32_F16 (fp16 in, fp32 acc),
// with pre-swizzled fp16 weight fragments, padded biases, branchless inner
// loops, and 32-bit offset addressing (saddr+voffset forms) throughout.
// ---------------------------------------------------------------------------

typedef _Float16 f16;
typedef __attribute__((ext_vector_type(16))) _Float16 v16h;
typedef __attribute__((ext_vector_type(2)))  _Float16 v2h;
typedef __attribute__((ext_vector_type(8)))  float    v8f;

#define NRES   60000
#define NGOD   300
#define NGODP  304          // god rows padded to multiple of 16
#define HEADS  4
#define HIDD   64
#define FDIM   256          // HEADS*HIDD
#define KCODES 256
#define ZDIM   32

// ------------------------------ utility kernels ----------------------------

__global__ void fillf_kernel(float* p, int n, float v) {
  int i = blockIdx.x * blockDim.x + threadIdx.x;
  if (i < n) p[i] = v;
}
__global__ void filli_kernel(int* p, int n, int v) {
  int i = blockIdx.x * blockDim.x + threadIdx.x;
  if (i < n) p[i] = v;
}
__global__ void padcopy_kernel(const float* __restrict__ src,
                               float* __restrict__ dst, int nsrc, int ntot) {
  int i = blockIdx.x * blockDim.x + threadIdx.x;
  if (i < ntot) dst[i] = (i < nsrc) ? src[i] : 0.0f;
}

__global__ void bn_kernel(const float* __restrict__ x,
                          const float* __restrict__ gamma,
                          const float* __restrict__ beta,
                          const float* __restrict__ mean,
                          const float* __restrict__ var,
                          float* __restrict__ out, int n) {
  int i = blockIdx.x * blockDim.x + threadIdx.x;
  if (i >= n) return;
  int f = i & 63;
  out[i] = (x[i] - mean[f]) / sqrtf(var[f] + 1e-5f) * gamma[f] + beta[f];
}

__global__ void scale_relu_kernel(const float* __restrict__ acc,
                                  float* __restrict__ h, int n, float scale) {
  int i = blockIdx.x * blockDim.x + threadIdx.x;
  if (i < n) h[i] = fmaxf(0.0f, scale * acc[i]);
}

// out[n,96] = concat(h[n,64], aa[n,20], zeros[12])
__global__ void concat_kernel(const float* __restrict__ h,
                              const float* __restrict__ aa,
                              float* __restrict__ out, int n) {
  int i = blockIdx.x * blockDim.x + threadIdx.x;
  if (i >= n) return;
  int node = i / 96, j = i % 96;
  out[i] = (j < 64) ? h[node * 64 + j]
         : (j < 84) ? aa[node * 20 + (j - 64)] : 0.0f;
}

// --------------------- weight pre-swizzle (B fragments) --------------------
// Bsw layout: [(nt*ksteps + ks)*32 + lane] * 16 halves, matching the CDNA5
// 32x16 fp16 B-fragment (lanes 0-15: K=0..7,16..23, lanes 16-31: K=8..15,24..31).
// Also emits the zero-padded bias (Np = ntiles*16 entries).
__global__ void prep_b_kernel(const float* __restrict__ W,
                              const float* __restrict__ bias, int K, int Nout,
                              int ksteps, int ntiles, f16* __restrict__ Bsw,
                              float* __restrict__ biasp) {
  int i = blockIdx.x * blockDim.x + threadIdx.x;
  int total = ntiles * ksteps * 32;
  if (i >= total) return;
  if (i < (ntiles << 4)) biasp[i] = (i < Nout) ? bias[i] : 0.0f;
  int lane = i & 31;
  int t = i >> 5;
  int ks = t % ksteps, nt = t / ksteps;
  int c = (nt << 4) + (lane & 15);
  int kb0 = ks * 32 + ((lane >= 16) ? 8 : 0);
  f16* out = Bsw + i * 16;
#pragma unroll
  for (int j = 0; j < 8; ++j) {
    int kb = kb0 + 2 * j + (j >= 4 ? 8 : 0);
    float b0 = (c < Nout && kb     < K) ? W[kb * Nout + c]       : 0.0f;
    float b1 = (c < Nout && kb + 1 < K) ? W[(kb + 1) * Nout + c] : 0.0f;
    out[2 * j]     = (f16)b0;
    out[2 * j + 1] = (f16)b1;
  }
}

// ------------------------------ WMMA GEMM ----------------------------------
// One wave computes a full 16 x (16*ntiles) strip of Out = act(A@W + bias).
// A fp32 (rows padded, K padded to 32), W pre-swizzled fp16 fragments,
// bias zero-padded. No bounds checks anywhere: everything is padded.
// All offsets are int32 -> compiler can use saddr + 32-bit voffset forms.
template <int ACT, bool ACCUM, bool OUTH, int KSTEPS>
__global__ __launch_bounds__(128)
void gemm_sw_kernel(const float* __restrict__ A, int lda, int Mtiles,
                    const f16* __restrict__ Bsw, int ntiles,
                    const float* __restrict__ biasp,
                    void* __restrict__ Out, int ldo) {
  int wv = (int)((blockIdx.x * blockDim.x + threadIdx.x) >> 5);
  if (wv >= Mtiles) return;
  const int lane = threadIdx.x & 31;
  const int m0 = wv << 4;
  const int r = m0 + (lane & 15);
  const bool hi = lane >= 16;

  // A fragments for all K-steps, kept in registers (KSTEPS is compile-time)
  v16h afr[KSTEPS];
  const int abase = r * lda + (hi ? 8 : 0);
#pragma unroll
  for (int ks = 0; ks < KSTEPS; ++ks) {
    const float* ap = A + abase + ks * 32;
    float4 x0 = *(const float4*)(ap);        // K+0..3
    float4 x1 = *(const float4*)(ap + 4);    // K+4..7
    float4 x2 = *(const float4*)(ap + 16);   // K+16..19
    float4 x3 = *(const float4*)(ap + 20);   // K+20..23
    v16h a;
    a[0]  = (_Float16)x0.x; a[1]  = (_Float16)x0.y;
    a[2]  = (_Float16)x0.z; a[3]  = (_Float16)x0.w;
    a[4]  = (_Float16)x1.x; a[5]  = (_Float16)x1.y;
    a[6]  = (_Float16)x1.z; a[7]  = (_Float16)x1.w;
    a[8]  = (_Float16)x2.x; a[9]  = (_Float16)x2.y;
    a[10] = (_Float16)x2.z; a[11] = (_Float16)x2.w;
    a[12] = (_Float16)x3.x; a[13] = (_Float16)x3.y;
    a[14] = (_Float16)x3.z; a[15] = (_Float16)x3.w;
    afr[ks] = a;
  }

  const int mb = m0 + (hi ? 8 : 0);
#pragma unroll 2
  for (int nt = 0; nt < ntiles; ++nt) {
    v8f acc = {};
    const v16h* bp = (const v16h*)Bsw + nt * (KSTEPS * 32) + lane;
#pragma unroll
    for (int ks = 0; ks < KSTEPS; ++ks) {
      v16h b = bp[ks * 32];
      acc = __builtin_amdgcn_wmma_f32_16x16x32_f16(false, afr[ks], false, b,
                                                   (short)0, acc, false, false);
    }
    int c = (nt << 4) + (lane & 15);
    float bv = biasp[c];                      // zero-padded: unconditional
    int ofs = mb * ldo + c;                   // 32-bit offsets -> saddr form
#pragma unroll
    for (int j = 0; j < 8; ++j, ofs += ldo) {
      float x = acc[j] + bv;
      if (ACCUM) x += ((float*)Out)[ofs];
      if (ACT == 1) x = x / (1.0f + __expf(-x));   // SiLU (silu(0)=0 in pads)
      if (OUTH) ((f16*)Out)[ofs] = (f16)x;
      else      ((float*)Out)[ofs] = x;
    }
  }
}

struct WInfo {
  const float* W;
  const float* b;
  int K, Nout, ksteps, ntiles;
  f16* sw;
  float* bp;
};

template <int ACT, bool ACCUM, bool OUTH>
static void run_gemm(const float* A, int lda, int M, const WInfo& w,
                     void* Out, int ldo, hipStream_t s) {
  int Mtiles = (M + 15) / 16;
  unsigned blocks = (unsigned)((Mtiles + 3) / 4);
  switch (w.ksteps) {
    case 1: gemm_sw_kernel<ACT, ACCUM, OUTH, 1><<<blocks, 128, 0, s>>>(
                A, lda, Mtiles, w.sw, w.ntiles, w.bp, Out, ldo); break;
    case 2: gemm_sw_kernel<ACT, ACCUM, OUTH, 2><<<blocks, 128, 0, s>>>(
                A, lda, Mtiles, w.sw, w.ntiles, w.bp, Out, ldo); break;
    case 3: gemm_sw_kernel<ACT, ACCUM, OUTH, 3><<<blocks, 128, 0, s>>>(
                A, lda, Mtiles, w.sw, w.ntiles, w.bp, Out, ldo); break;
    default: gemm_sw_kernel<ACT, ACCUM, OUTH, 4><<<blocks, 128, 0, s>>>(
                A, lda, Mtiles, w.sw, w.ntiles, w.bp, Out, ldo); break;
  }
}

// --------------------------- edge attention passes -------------------------

__device__ inline float dot64(const f16* __restrict__ a,
                              const f16* __restrict__ b) {
  const v16h* av = (const v16h*)a;
  const v16h* bv = (const v16h*)b;
  float acc = 0.0f;
#pragma unroll
  for (int t = 0; t < 4; ++t) {
    v16h x = av[t], y = bv[t];
#pragma unroll
    for (int j = 0; j < 8; ++j) {
      v2h xp = {x[2 * j], x[2 * j + 1]};
      v2h yp = {y[2 * j], y[2 * j + 1]};
#if __has_builtin(__builtin_amdgcn_fdot2)
      acc = __builtin_amdgcn_fdot2(xp, yp, acc, false);  // v_dot2_f32_f16
#else
      acc += (float)xp[0] * (float)yp[0] + (float)xp[1] * (float)yp[1];
#endif
    }
  }
  return acc;
}

__device__ inline void atomicMaxFloat(float* addr, float val) {
  unsigned int* ua = (unsigned int*)addr;
  unsigned int old = *ua;
  while (__uint_as_float(old) < val) {
    unsigned int assumed = old;
    old = atomicCAS(ua, assumed, __float_as_uint(val));
    if (old == assumed) break;
  }
}

// pass 1: logits[e,h] = <q[dst,h,:],k[src,h,:]>/8 ; segmax over dst
__global__ void edge_logits_kernel(const f16* __restrict__ q,
                                   const f16* __restrict__ k,
                                   const int* __restrict__ src,
                                   const int* __restrict__ dst, int E,
                                   float* __restrict__ logits,
                                   float* __restrict__ segmax) {
  int i = blockIdx.x * blockDim.x + threadIdx.x;
  if (i >= E * HEADS) return;
  int e = i >> 2, h = i & 3;
  int s = src[e], d = dst[e];
  float lg = dot64(q + d * FDIM + h * HIDD,
                   k + s * FDIM + h * HIDD) * 0.125f;
  logits[i] = lg;
  atomicMaxFloat(&segmax[d * HEADS + h], lg);
}

// pass 2: w = exp(l - m[dst]); segsum += w (w overwrites logits)
__global__ void edge_expsum_kernel(const int* __restrict__ dst, int E,
                                   float* __restrict__ logits,
                                   const float* __restrict__ segmax,
                                   float* __restrict__ segsum) {
  int i = blockIdx.x * blockDim.x + threadIdx.x;
  if (i >= E * HEADS) return;
  int e = i >> 2, h = i & 3;
  int d = dst[e];
  float w = __expf(logits[i] - segmax[d * HEADS + h]);
  logits[i] = w;
  atomicAdd(&segsum[d * HEADS + h], w);
}

// pass 3: one wave per edge; head-mean folded in (x0.25); 2 dims per lane
__global__ void edge_accum_kernel(const f16* __restrict__ v,
                                  const int* __restrict__ src,
                                  const int* __restrict__ dst, int E,
                                  const float* __restrict__ w,
                                  const float* __restrict__ segsum,
                                  float* __restrict__ acc) {
  int e = (blockIdx.x * blockDim.x + threadIdx.x) >> 5;
  int lane = threadIdx.x & 31;
  if (e >= E) return;
  int s = src[e], d = dst[e];
  int d0 = lane * 2, d1 = lane * 2 + 1;
  float s0 = 0.0f, s1 = 0.0f;
#pragma unroll
  for (int h = 0; h < HEADS; ++h) {
    float alpha = w[e * HEADS + h] / (segsum[d * HEADS + h] + 1e-16f) * 0.25f;
    s0 += alpha * (float)v[s * FDIM + h * HIDD + d0];
    s1 += alpha * (float)v[s * FDIM + h * HIDD + d1];
  }
  atomicAdd(&acc[d * HIDD + d0], s0);
  atomicAdd(&acc[d * HIDD + d1], s1);
}

// godnode conv: block per god (200 residues fan-in), no atomics
__global__ __launch_bounds__(256)
void god_attn_kernel(const f16* __restrict__ qg, const f16* __restrict__ kb,
                     const f16* __restrict__ vb, const int* __restrict__ src,
                     float* __restrict__ gacc /* [NGODP,64], holds skip */) {
  __shared__ float slog[200 * HEADS];
  __shared__ float smax[HEADS], ssum[HEADS];
  __shared__ float part[HEADS][HIDD];
  int g = blockIdx.x, t = threadIdx.x;
  for (int i = t; i < 200 * HEADS; i += 256) {
    int e = g * 200 + (i >> 2), h = i & 3;
    int s = src[e];
    slog[i] = dot64(qg + g * FDIM + h * HIDD,
                    kb + s * FDIM + h * HIDD) * 0.125f;
  }
  __syncthreads();
  if (t < HEADS) {
    float m = -3.0e38f;
    for (int e = 0; e < 200; ++e) m = fmaxf(m, slog[e * HEADS + t]);
    smax[t] = m;
  }
  __syncthreads();
  for (int i = t; i < 200 * HEADS; i += 256)
    slog[i] = __expf(slog[i] - smax[i & 3]);
  __syncthreads();
  if (t < HEADS) {
    float s = 0.0f;
    for (int e = 0; e < 200; ++e) s += slog[e * HEADS + t];
    ssum[t] = s + 1e-16f;
  }
  __syncthreads();
  int h = t >> 6, dim = t & 63;
  float a = 0.0f;
  for (int e = 0; e < 200; ++e) {
    int s = src[g * 200 + e];
    a += slog[e * HEADS + h] * (float)vb[s * FDIM + h * HIDD + dim];
  }
  part[h][dim] = a / ssum[h];
  __syncthreads();
  if (t < HIDD)
    gacc[g * HIDD + t] +=
        0.25f * (part[0][t] + part[1][t] + part[2][t] + part[3][t]);
}

// ------------------------------- VQ stage ----------------------------------

__global__ __launch_bounds__(256)
void vq_kernel(const float* __restrict__ z, const float* __restrict__ emb,
               float* __restrict__ out, int* __restrict__ counts,
               float* __restrict__ msesum, int N) {
  __shared__ float semb[KCODES * ZDIM];     // 32 KB codebook in LDS
  for (int i = threadIdx.x; i < KCODES * ZDIM; i += blockDim.x)
    semb[i] = emb[i];
  __syncthreads();
  int n = blockIdx.x * blockDim.x + threadIdx.x;
  if (n >= N) return;
  float zv[ZDIM];
#pragma unroll
  for (int j = 0; j < ZDIM; ++j) zv[j] = z[n * ZDIM + j];
  float best = 3.4e38f; int bidx = 0;
  for (int c = 0; c < KCODES; ++c) {
    float ee = 0.0f, ze = 0.0f;
#pragma unroll
    for (int j = 0; j < ZDIM; ++j) {
      float e = semb[c * ZDIM + j];
      ee += e * e; ze += zv[j] * e;
    }
    float dist = ee - 2.0f * ze;            // + |z|^2 const for argmin
    if (dist < best) { best = dist; bidx = c; }
  }
  float mse = 0.0f;
#pragma unroll
  for (int j = 0; j < ZDIM; ++j) {
    float qv = semb[bidx * ZDIM + j];
    out[n * ZDIM + j] = qv;                 // q_st == quant numerically
    float d = qv - zv[j];
    mse += d * d;
  }
  atomicAdd(msesum, mse);
  atomicAdd(&counts[bidx], 1);
}

__global__ __launch_bounds__(256)
void finalize_kernel(const int* __restrict__ counts,
                     const float* __restrict__ msesum,
                     float* __restrict__ out, int N) {
  __shared__ float red[KCODES];
  int t = threadIdx.x;
  float p = (float)counts[t] / (float)N;
  float c = p * logf(p + 1e-10f) +                         // -entropy
            (p - 1.0f / KCODES) * (p - 1.0f / KCODES) +    // diversity
            p * logf(p * (float)KCODES + 1e-10f);          // kl
  red[t] = c;
  __syncthreads();
  for (int s = 128; s > 0; s >>= 1) {
    if (t < s) red[t] += red[t + s];
    __syncthreads();
  }
  if (t == 0)
    out[0] = 1.25f * (msesum[0] / (float)(N * ZDIM)) + red[0];
}

// ------------------------------- launcher ----------------------------------

extern "C" void kernel_launch(void* const* d_in, const int* in_sizes, int n_in,
                              void* d_out, int out_size, void* d_ws,
                              size_t ws_size, hipStream_t stream) {
  const float* x_res = (const float*)d_in[0];
  const float* x_god = (const float*)d_in[1];
  const float* x_aa  = (const float*)d_in[2];
  const int* ei_bb   = (const int*)d_in[3];
  const int* ei_cp   = (const int*)d_in[4];
  const int* ei_inf  = (const int*)d_in[5];
  const float* bn_g  = (const float*)d_in[6];
  const float* bn_b  = (const float*)d_in[7];
  const float* bn_m  = (const float*)d_in[8];
  const float* bn_v  = (const float*)d_in[9];
  auto LW = [&](int L, int cv, int l) {
    return (const float*)d_in[10 + L * 24 + cv * 8 + l * 2];
  };
  auto LB = [&](int L, int cv, int l) {
    return (const float*)d_in[10 + L * 24 + cv * 8 + l * 2 + 1];
  };
  auto DW = [&](int j) { return (const float*)d_in[58 + 2 * j]; };
  auto DB = [&](int j) { return (const float*)d_in[59 + 2 * j]; };
  const float* emb = (const float*)d_in[74];

  const int E_bb = in_sizes[3] / 2;
  const int E_cp = in_sizes[4] / 2;

  // ----- workspace layout -----
  size_t off = 0;
  auto alloc = [&](size_t bytes) {
    size_t o = off;
    off = (off + bytes + 255) & ~(size_t)255;
    return o;
  };
  char* B = (char*)d_ws;
  float* hA     = (float*)(B + alloc((size_t)NRES * HIDD * 4));
  float* acc    = (float*)(B + alloc((size_t)NRES * HIDD * 4));
  float* g_in   = (float*)(B + alloc((size_t)NGODP * 32 * 4));
  float* gA     = (float*)(B + alloc((size_t)NGODP * HIDD * 4));
  float* gacc   = (float*)(B + alloc((size_t)NGODP * HIDD * 4));
  float* segmax = (float*)(B + alloc((size_t)NRES * HEADS * 4));
  float* segsum = (float*)(B + alloc((size_t)NRES * HEADS * 4));
  float* logits = (float*)(B + alloc((size_t)E_cp * HEADS * 4));
  f16*   qbuf   = (f16*)  (B + alloc((size_t)NRES * FDIM * 2));
  f16*   kbuf   = (f16*)  (B + alloc((size_t)NRES * FDIM * 2));
  f16*   vbuf   = (f16*)  (B + alloc((size_t)NRES * FDIM * 2));
  f16*   qg     = (f16*)  (B + alloc((size_t)NGODP * FDIM * 2));
  int*   counts = (int*)  (B + alloc(KCODES * 4));
  float* msesum = (float*)(B + alloc(256));
  // MLP ping-pong reuses the (idle) q/k/v regions after the GNN stage
  float* dq = (float*)qbuf;   // concat [N,96] then odd-layer outputs [N,128]
  float* dk = (float*)kbuf;   // even-layer outputs [N,128]
  float* zb = (float*)vbuf;   // final z [N,32]

  // ----- weight swizzle table (24 GNN + 8 dense) -----
  const int dims[9] = {84, 100, 100, 100, 100, 100, 100, 100, 32};
  WInfo wi[32];
  for (int L = 0; L < 2; ++L)
    for (int cv = 0; cv < 3; ++cv)
      for (int l = 0; l < 4; ++l) {
        int dg = (L == 0) ? 32 : 64;
        int K = (cv == 2 && (l == 0 || l == 3)) ? dg : HIDD;  // inf q/skip
        int Nout = (l == 3) ? HIDD : FDIM;
        wi[L * 12 + cv * 4 + l] = {LW(L, cv, l), LB(L, cv, l), K, Nout,
                                   0, 0, nullptr, nullptr};
      }
  for (int j = 0; j < 8; ++j)
    wi[24 + j] = {DW(j), DB(j), dims[j], dims[j + 1], 0, 0, nullptr, nullptr};
  for (int i = 0; i < 32; ++i) {
    wi[i].ksteps = (wi[i].K + 31) / 32;
    int Np = (wi[i].Nout + 31) & ~31;          // pad N to 32 => next K pad ok
    wi[i].ntiles = Np / 16;
    wi[i].sw = (f16*)(B + alloc((size_t)wi[i].ntiles * wi[i].ksteps * 32 *
                                16 * sizeof(f16)));
    wi[i].bp = (float*)(B + alloc((size_t)Np * sizeof(float)));
  }

  auto grid1 = [](long n) { return (unsigned)((n + 255) / 256); };

  // ----- pre-swizzle all weights to fp16 fragments + padded biases -----
  for (int i = 0; i < 32; ++i) {
    int total = wi[i].ntiles * wi[i].ksteps * 32;
    prep_b_kernel<<<grid1(total), 256, 0, stream>>>(
        wi[i].W, wi[i].b, wi[i].K, wi[i].Nout, wi[i].ksteps, wi[i].ntiles,
        wi[i].sw, wi[i].bp);
  }

  // ----- BN + god input padding -----
  bn_kernel<<<grid1((long)NRES * HIDD), 256, 0, stream>>>(
      x_res, bn_g, bn_b, bn_m, bn_v, hA, NRES * HIDD);
  padcopy_kernel<<<grid1((long)NGODP * 32), 256, 0, stream>>>(
      x_god, g_in, NGOD * 32, NGODP * 32);

  const float* gin = g_in;
  int glda = 32;
  for (int L = 0; L < 2; ++L) {
    for (int cv = 0; cv < 2; ++cv) {             // 0 = bb, 1 = cp
      const int* src = (cv == 0) ? ei_bb : ei_cp;
      const int E    = (cv == 0) ? E_bb : E_cp;
      const int* dst = src + E;
      if (cv == 0)
        run_gemm<0, false, false>(hA, HIDD, NRES, wi[L * 12 + cv * 4 + 3],
                                  acc, HIDD, stream);
      else
        run_gemm<0, true, false>(hA, HIDD, NRES, wi[L * 12 + cv * 4 + 3],
                                 acc, HIDD, stream);
      run_gemm<0, false, true>(hA, HIDD, NRES, wi[L * 12 + cv * 4 + 0],
                               qbuf, FDIM, stream);
      run_gemm<0, false, true>(hA, HIDD, NRES, wi[L * 12 + cv * 4 + 1],
                               kbuf, FDIM, stream);
      run_gemm<0, false, true>(hA, HIDD, NRES, wi[L * 12 + cv * 4 + 2],
                               vbuf, FDIM, stream);
      fillf_kernel<<<grid1((long)NRES * HEADS), 256, 0, stream>>>(
          segmax, NRES * HEADS, -3.0e38f);
      fillf_kernel<<<grid1((long)NRES * HEADS), 256, 0, stream>>>(
          segsum, NRES * HEADS, 0.0f);
      edge_logits_kernel<<<grid1((long)E * HEADS), 256, 0, stream>>>(
          qbuf, kbuf, src, dst, E, logits, segmax);
      edge_expsum_kernel<<<grid1((long)E * HEADS), 256, 0, stream>>>(
          dst, E, logits, segmax, segsum);
      edge_accum_kernel<<<grid1((long)E * 32), 256, 0, stream>>>(
          vbuf, src, dst, E, logits, segsum, acc);
    }
    // inf conv: residues -> godnodes
    run_gemm<0, false, false>(gin, glda, NGODP, wi[L * 12 + 2 * 4 + 3],
                              gacc, HIDD, stream);
    run_gemm<0, false, true>(gin, glda, NGODP, wi[L * 12 + 2 * 4 + 0],
                             qg, FDIM, stream);
    run_gemm<0, false, true>(hA, HIDD, NRES, wi[L * 12 + 2 * 4 + 1],
                             kbuf, FDIM, stream);
    run_gemm<0, false, true>(hA, HIDD, NRES, wi[L * 12 + 2 * 4 + 2],
                             vbuf, FDIM, stream);
    god_attn_kernel<<<NGOD, 256, 0, stream>>>(qg, kbuf, vbuf, ei_inf, gacc);
    scale_relu_kernel<<<grid1((long)NRES * HIDD), 256, 0, stream>>>(
        acc, hA, NRES * HIDD, 0.5f);
    scale_relu_kernel<<<grid1((long)NGODP * HIDD), 256, 0, stream>>>(
        gacc, gA, NGODP * HIDD, 1.0f);
    gin = gA;
    glda = HIDD;
  }

  // ----- dense SiLU stack: 84(->96) -> 100(->128) x7 -> 32 -----
  concat_kernel<<<grid1((long)NRES * 96), 256, 0, stream>>>(
      hA, x_aa, dq, NRES * 96);
  const float* cur = dq;
  int curld = 96;
  float* bufs[2] = {dq, dk};
  for (int j = 0; j < 8; ++j) {
    float* outb = (j == 7) ? zb : bufs[(j + 1) & 1];
    int outld = (j == 7) ? 32 : 128;
    run_gemm<1, false, false>(cur, curld, NRES, wi[24 + j], outb, outld,
                              stream);
    cur = outb;
    curld = outld;
  }

  // ----- VQ + losses -----
  filli_kernel<<<1, 256, 0, stream>>>(counts, KCODES, 0);
  fillf_kernel<<<1, 32, 0, stream>>>(msesum, 1, 0.0f);
  float* outf = (float*)d_out;
  vq_kernel<<<grid1(NRES), 256, 0, stream>>>(zb, emb, outf, counts, msesum,
                                             NRES);
  finalize_kernel<<<1, 256, 0, stream>>>(counts, msesum,
                                         outf + NRES * ZDIM, NRES);
}
